// HMF_Block_55190329754131
// MI455X (gfx1250) — matrix operand order repaired and checked by reading, not compiled, not run on hardware
//
#include <hip/hip_runtime.h>
#include <hip/hip_bf16.h>

typedef __attribute__((ext_vector_type(16))) _Float16 v16h;
typedef __attribute__((ext_vector_type(8)))  _Float16 v8h;
typedef __attribute__((ext_vector_type(8)))  float    v8f;

#define BB   4
#define CC   128
#define HH   96
#define WW   320
#define CM   32
#define DD   32
#define HWP  (HH * WW)          // 30720 pixels per batch image
#define KRD  (CC * 9)           // 1152, implicit-GEMM K for 3x3 conv

__device__ __forceinline__ float sigmoidf_(float x) { return 1.0f / (1.0f + __expf(-x)); }

__device__ __forceinline__ v16h cat16(v8h lo, v8h hi) {
    return __builtin_shufflevector(lo, hi, 0, 1, 2, 3, 4, 5, 6, 7,
                                   8, 9, 10, 11, 12, 13, 14, 15);
}

// ---------------------------------------------------------------------------
// Kernel 0: one-time repack of w_rd [O][C][3][3] f32 -> w_h [O][tap][C] f16.
// Makes implicit-GEMM A fragments two contiguous 16B loads per lane.
// ---------------------------------------------------------------------------
__global__ __launch_bounds__(256)
void wrd_pack_kernel(const float* __restrict__ w_rd, _Float16* __restrict__ w_h)
{
    const int total = CC * 9 * CC;   // 147456
    for (int i = blockIdx.x * 256 + threadIdx.x; i < total; i += gridDim.x * 256) {
        int o   = i / (9 * CC);
        int r   = i % (9 * CC);
        int tap = r / CC;
        int c   = r % CC;
        w_h[i] = (_Float16)w_rd[(o * CC + c) * 9 + tap];
    }
}

// ---------------------------------------------------------------------------
// Kernel 1: q/k = l2norm_c(conv1x1(feat, W, b)).  GEMM [32 x 128] @ [128 x N]
// via WMMA f32_16x16x32_f16, operands staged as packed f16 in LDS.
// Block = 256 thr (8 waves) = 2 m-tiles x 4 n-tiles, 64 pixels per block.
// ---------------------------------------------------------------------------
__global__ __launch_bounds__(256)
void qk_norm_kernel(const float* __restrict__ feat, const float* __restrict__ Wm,
                    const float* __restrict__ bias, float* __restrict__ out)
{
    __shared__ __align__(32) _Float16 lds_xh[64][CC];   // [pixel][channel]
    __shared__ __align__(32) _Float16 lds_wh[CM][CC];   // [out-ch][channel]
    __shared__ float lds_o[CM][64];
    __shared__ float lds_b[CM];
    __shared__ float inv_n[64];

    const int tid  = threadIdx.x;
    const int b    = blockIdx.x / 480;
    const int pix0 = (blockIdx.x % 480) * 64;
    const float* fb = feat + (size_t)b * CC * HWP;

    for (int idx = tid; idx < CC * 64; idx += 256) {
        int p = idx & 63, c = idx >> 6;               // coalesced over p
        lds_xh[p][c] = (_Float16)fb[(size_t)c * HWP + pix0 + p];
    }
    for (int idx = tid; idx < CM * CC; idx += 256)
        lds_wh[idx >> 7][idx & 127] = (_Float16)Wm[idx];
    if (tid < CM) lds_b[tid] = bias[tid];
    __syncthreads();

    const int wid  = tid >> 5;
    const int lane = tid & 31;
    const int hl   = lane & 15;
    const int hi   = lane >> 4;
    const int mt   = (wid & 1) * 16;
    const int nt   = (wid >> 1) * 16;

    v8f acc = {};
#pragma unroll
    for (int kc = 0; kc < CC; kc += 32) {
        v8h alo = *(const v8h*)&lds_wh[mt + hl][kc + (hi ? 16 : 0)];
        v8h ahi = *(const v8h*)&lds_wh[mt + hl][kc + (hi ? 24 : 8)];
        v16h a  = cat16(alo, ahi);
        v16h bf = *(const v16h*)&lds_xh[nt + hl][kc + (hi ? 16 : 0)];
        acc = __builtin_amdgcn_wmma_f32_16x16x32_f16(false, a, false, bf,
                                                     (short)0, acc, false, false);
    }
#pragma unroll
    for (int i = 0; i < 8; ++i) {
        int m = mt + i + (hi ? 8 : 0);
        lds_o[m][nt + hl] = acc[i] + lds_b[m];
    }
    __syncthreads();

    if (tid < 64) {
        float s = 0.f;
#pragma unroll
        for (int c = 0; c < CM; ++c) { float v = lds_o[c][tid]; s += v * v; }
        inv_n[tid] = 1.0f / fmaxf(sqrtf(s), 1e-12f);
    }
    __syncthreads();

    float* ob = out + (size_t)b * CM * HWP;
    for (int idx = tid; idx < CM * 64; idx += 256) {
        int c = idx >> 6, p = idx & 63;
        ob[(size_t)c * HWP + pix0 + p] = lds_o[c][p] * inv_n[p];
    }
}

// ---------------------------------------------------------------------------
// Kernel 2: fused warp + metric matvec + sim/tanh + k_mean + disparity gate.
// One block per (b,h) row; 320 threads = one pixel each.  k row staged in LDS;
// warped volume (503 MB!) never materialized.
// ---------------------------------------------------------------------------
__global__ __launch_bounds__(320)
void warp_sim_kernel(const float* __restrict__ q, const float* __restrict__ k,
                     const float* __restrict__ metric, const float* __restrict__ disps,
                     const float* __restrict__ directs, const float* __restrict__ alpha_p,
                     const float* __restrict__ beta_p, const float* __restrict__ w_da,
                     const float* __restrict__ b_da, float* __restrict__ Cv,
                     float* __restrict__ da_out)
{
    __shared__ float lds_k[CM][WW];
    __shared__ float lds_M[CM][CM];
    __shared__ float lds_wda[DD][2 * CM];
    __shared__ float lds_disp[DD];

    const int tid = threadIdx.x;
    const int b   = blockIdx.x / HH;
    const int h   = blockIdx.x % HH;

    for (int idx = tid; idx < CM * WW; idx += 320) {
        int c = idx / WW, w = idx % WW;
        lds_k[c][w] = k[((size_t)(b * CM + c) * HH + h) * WW + w];
    }
    for (int idx = tid; idx < CM * CM; idx += 320) {
        int c = idx >> 5, d = idx & 31;
        lds_M[c][d] = 0.5f * (metric[c * CM + d] + metric[d * CM + c]);
    }
    for (int idx = tid; idx < DD * 2 * CM; idx += 320)
        lds_wda[idx / (2 * CM)][idx % (2 * CM)] = w_da[idx];
    if (tid < DD) lds_disp[tid] = disps[tid];
    __syncthreads();

    const int   w      = tid;                // 320 threads == W
    const float alpha  = alpha_p[0];
    const float beta   = beta_p[0];
    const float direct = directs[b];

    float qreg[CM], qproj[CM], kmean[CM];
#pragma unroll
    for (int c = 0; c < CM; ++c) {
        qreg[c]  = q[((size_t)(b * CM + c) * HH + h) * WW + w];
        kmean[c] = 0.f;
    }
#pragma unroll
    for (int d = 0; d < CM; ++d) {
        float s = 0.f;
#pragma unroll
        for (int c = 0; c < CM; ++c) s += qreg[c] * lds_M[c][d];
        qproj[d] = s;
    }

    for (int d = 0; d < DD; ++d) {
        float shift = lds_disp[d] * direct * (float)(WW - 1);
        float px = fminf(fmaxf((float)w + shift, 0.0f), (float)(WW - 1));
        int   x0 = (int)floorf(px);
        x0 = min(max(x0, 0), WW - 1);
        int   x1 = min(x0 + 1, WW - 1);
        float f  = px - (float)x0;
        float sim = 0.f;
#pragma unroll
        for (int c = 0; c < CM; ++c) {
            float wv = (1.0f - f) * lds_k[c][x0] + f * lds_k[c][x1];
            sim      += qproj[c] * wv;
            kmean[c] += wv;
        }
        Cv[((size_t)(b * DD + d) * HH + h) * WW + w] = tanhf(alpha * sim + beta);
    }
#pragma unroll
    for (int c = 0; c < CM; ++c) kmean[c] *= (1.0f / (float)DD);

    for (int d = 0; d < DD; ++d) {
        float s = b_da[d];
#pragma unroll
        for (int c = 0; c < CM; ++c)
            s += lds_wda[d][c] * qreg[c] + lds_wda[d][CM + c] * kmean[c];
        da_out[((size_t)(b * DD + d) * HH + h) * WW + w] = sigmoidf_(s);
    }
}

// ---------------------------------------------------------------------------
// Kernel 3: per-slice 3x3 conv (SAME, zero pad) * da, softmax over D -> P.
// ---------------------------------------------------------------------------
__global__ __launch_bounds__(256)
void cf_softmax_kernel(const float* __restrict__ Cv, const float* __restrict__ da,
                       const float* __restrict__ w_cf, const float* __restrict__ b_cf,
                       float* __restrict__ P)
{
    const int tid = threadIdx.x;
    const int b   = blockIdx.x / HH;
    const int h   = blockIdx.x % HH;

    float wcf[9];
#pragma unroll
    for (int i = 0; i < 9; ++i) wcf[i] = w_cf[i];
    const float bcf = b_cf[0];

    for (int w = tid; w < WW; w += 256) {
        float vals[DD];
        float mx = -1e30f;
#pragma unroll
        for (int d = 0; d < DD; ++d) {
            const float* cvp = Cv + (size_t)(b * DD + d) * HWP;
            float s = bcf;
#pragma unroll
            for (int ky = 0; ky < 3; ++ky) {
                int hh = h + ky - 1;
                if (hh >= 0 && hh < HH) {
#pragma unroll
                    for (int kx = 0; kx < 3; ++kx) {
                        int ww = w + kx - 1;
                        if (ww >= 0 && ww < WW) s += wcf[ky * 3 + kx] * cvp[hh * WW + ww];
                    }
                }
            }
            s *= da[((size_t)(b * DD + d) * HH + h) * WW + w];
            vals[d] = s;
            mx = fmaxf(mx, s);
        }
        float den = 0.f;
#pragma unroll
        for (int d = 0; d < DD; ++d) { vals[d] = __expf(vals[d] - mx); den += vals[d]; }
        float inv = 1.0f / den;
#pragma unroll
        for (int d = 0; d < DD; ++d)
            P[((size_t)(b * DD + d) * HH + h) * WW + w] = vals[d] * inv;
    }
}

// ---------------------------------------------------------------------------
// Kernel 4: cost_feat = w_ex[128x32] @ P (WMMA, K=32, f16 staged) + b_ex, then
// g = sigmoid(w_fg . [t_feat; cost_feat] + b_fg), fused = g*t + (1-g)*cost.
// ---------------------------------------------------------------------------
__global__ __launch_bounds__(256)
void cost_fuse_kernel(const float* __restrict__ P, const float* __restrict__ t_feat,
                      const float* __restrict__ w_ex, const float* __restrict__ b_ex,
                      const float* __restrict__ w_fg, const float* __restrict__ b_fg,
                      float* __restrict__ fused)
{
    __shared__ __align__(32) _Float16 lds_Ph[16][DD];    // [pixel][d]
    __shared__ __align__(32) _Float16 lds_wexh[CC][DD];  // [out-ch][d]
    __shared__ float lds_cost[CC][16];
    __shared__ float lds_part[16][16];
    __shared__ float lds_g[16];
    __shared__ float lds_bex[CC];

    const int tid  = threadIdx.x;
    const int b    = blockIdx.x / 1920;
    const int pix0 = (blockIdx.x % 1920) * 16;

    for (int idx = tid; idx < DD * 16; idx += 256) {
        int d = idx >> 4, n = idx & 15;               // coalesced over n
        lds_Ph[n][d] = (_Float16)P[(size_t)(b * DD + d) * HWP + pix0 + n];
    }
    for (int idx = tid; idx < CC * DD; idx += 256)
        lds_wexh[idx >> 5][idx & 31] = (_Float16)w_ex[idx];
    if (tid < CC) lds_bex[tid] = b_ex[tid];
    __syncthreads();

    const int wid  = tid >> 5;
    const int lane = tid & 31;
    const int hl   = lane & 15;
    const int hi   = lane >> 4;
    const int m0   = wid * 16;

    v8h alo = *(const v8h*)&lds_wexh[m0 + hl][hi ? 16 : 0];
    v8h ahi = *(const v8h*)&lds_wexh[m0 + hl][hi ? 24 : 8];
    v16h a  = cat16(alo, ahi);
    v16h bf = *(const v16h*)&lds_Ph[hl][hi ? 16 : 0];
    v8f acc = {};
    acc = __builtin_amdgcn_wmma_f32_16x16x32_f16(false, a, false, bf,
                                                 (short)0, acc, false, false);
#pragma unroll
    for (int i = 0; i < 8; ++i) {
        int o = m0 + i + (hi ? 8 : 0);
        lds_cost[o][hl] = acc[i] + lds_bex[o];
    }
    __syncthreads();

    { // gate partial sums: 16 threads per pixel, 8 channels each
        int n = tid & 15, cp = tid >> 4;
        const float* tb = t_feat + (size_t)(b * CC) * HWP + pix0 + n;
        float s = 0.f;
#pragma unroll
        for (int j = 0; j < 8; ++j) {
            int c = cp * 8 + j;
            s += w_fg[c] * tb[(size_t)c * HWP] + w_fg[CC + c] * lds_cost[c][n];
        }
        lds_part[cp][n] = s;
    }
    __syncthreads();
    if (tid < 16) {
        float g = b_fg[0];
#pragma unroll
        for (int i = 0; i < 16; ++i) g += lds_part[i][tid];
        lds_g[tid] = sigmoidf_(g);
    }
    __syncthreads();

    for (int idx = tid; idx < CC * 16; idx += 256) {
        int c = idx >> 4, n = idx & 15;
        float tf = t_feat[(size_t)(b * CC + c) * HWP + pix0 + n];
        float g  = lds_g[n];
        fused[(size_t)(b * CC + c) * HWP + pix0 + n] = g * tf + (1.f - g) * lds_cost[c][n];
    }
}

// ---------------------------------------------------------------------------
// Kernel 5: C->C 3x3 conv as implicit GEMM (M=128, K=9*128, N=pixels), WMMA.
// K order = tap*128 + c; weights pre-packed f16 [O][tap][c]; input tile staged
// as f16 [3][34][128] (channel-contiguous).  Block = 32 pixels x 128 out-ch,
// 8 waves, each wave: 1 m-tile x 2 n-tiles, A fragment reused -> 72 WMMA/wave.
// ---------------------------------------------------------------------------
__global__ __launch_bounds__(256)
void conv3x3_wmma_kernel(const float* __restrict__ fused,
                         const _Float16* __restrict__ w_h,
                         const float* __restrict__ b_rd, float* __restrict__ xout)
{
    __shared__ __align__(32) _Float16 lds_in[3][34][CC];  // [ky][x][c]

    const int tid = threadIdx.x;
    const int b   = blockIdx.x / (HH * 10);
    const int rem = blockIdx.x % (HH * 10);
    const int h   = rem / 10;
    const int w0  = (rem % 10) * 32;

    for (int idx = tid; idx < CC * 3 * 34; idx += 256) {
        int c  = idx / (3 * 34);
        int r  = idx % (3 * 34);
        int ky = r / 34, xx = r % 34;                 // coalesced over xx
        int hh = h + ky - 1, ww = w0 + xx - 1;
        float v = 0.f;
        if (hh >= 0 && hh < HH && ww >= 0 && ww < WW)
            v = fused[((size_t)(b * CC + c) * HH + hh) * WW + ww];
        lds_in[ky][xx][c] = (_Float16)v;
    }
    __syncthreads();

    const int wid  = tid >> 5;
    const int lane = tid & 31;
    const int hl   = lane & 15;
    const int hi   = lane >> 4;
    const int m0   = wid * 16;
    const _Float16* wrowh = w_h + (size_t)(m0 + hl) * KRD;
    const int koffa0 = hi ? 16 : 0;
    const int koffa1 = hi ? 24 : 8;
    const int koffb  = hi ? 16 : 0;

    v8f acc0 = {}, acc1 = {};
    for (int tap = 0; tap < 9; ++tap) {
        const int ky = tap / 3, kx = tap % 3;
        const _Float16* brow0 = &lds_in[ky][hl + kx][0];
        const _Float16* brow1 = &lds_in[ky][hl + 16 + kx][0];
        const _Float16* wtap  = wrowh + tap * CC;
        __builtin_prefetch(wtap + CC, 0, 1);          // global_prefetch next tap
#pragma unroll
        for (int cchunk = 0; cchunk < CC; cchunk += 32) {
            v8h alo = *(const v8h*)(wtap + cchunk + koffa0);
            v8h ahi = *(const v8h*)(wtap + cchunk + koffa1);
            v16h a  = cat16(alo, ahi);
            v16h b0 = *(const v16h*)(brow0 + cchunk + koffb);
            v16h b1 = *(const v16h*)(brow1 + cchunk + koffb);
            acc0 = __builtin_amdgcn_wmma_f32_16x16x32_f16(false, a, false, b0,
                                                          (short)0, acc0, false, false);
            acc1 = __builtin_amdgcn_wmma_f32_16x16x32_f16(false, a, false, b1,
                                                          (short)0, acc1, false, false);
        }
    }
#pragma unroll
    for (int i = 0; i < 8; ++i) {
        int o = m0 + i + (hi ? 8 : 0);
        float bias = b_rd[o];
        size_t base = ((size_t)(b * CC + o) * HH + h) * WW + w0;
        xout[base + hl]      = acc0[i] + bias;
        xout[base + 16 + hl] = acc1[i] + bias;
    }
}

// ---------------------------------------------------------------------------
// Kernel 6-8: SE layer (global avg pool -> FC/relu -> FC/sigmoid -> scale+ELU)
// ---------------------------------------------------------------------------
__global__ __launch_bounds__(256)
void se_reduce_kernel(const float* __restrict__ x, float* __restrict__ y)
{
    __shared__ float red[256];
    const int bc = blockIdx.x;                 // 0..B*C-1
    const float* p = x + (size_t)bc * HWP;
    float s = 0.f;
    for (int i = threadIdx.x; i < HWP; i += 256) s += p[i];
    red[threadIdx.x] = s;
    __syncthreads();
    for (int st = 128; st > 0; st >>= 1) {
        if (threadIdx.x < st) red[threadIdx.x] += red[threadIdx.x + st];
        __syncthreads();
    }
    if (threadIdx.x == 0) y[bc] = red[0] / (float)HWP;
}

__global__ __launch_bounds__(256)
void se_fc_kernel(const float* __restrict__ y, const float* __restrict__ w1,
                  const float* __restrict__ w2, float* __restrict__ sc)
{
    __shared__ float hid[BB][8];
    const int t = threadIdx.x;
    if (t < BB * 8) {
        int b = t >> 3, j = t & 7;
        float s = 0.f;
        for (int c = 0; c < CC; ++c) s += w1[j * CC + c] * y[b * CC + c];
        hid[b][j] = fmaxf(s, 0.f);
    }
    __syncthreads();
    for (int idx = t; idx < BB * CC; idx += 256) {
        int b = idx >> 7, c = idx & 127;
        float s = 0.f;
#pragma unroll
        for (int j = 0; j < 8; ++j) s += w2[c * 8 + j] * hid[b][j];
        sc[idx] = sigmoidf_(s);
    }
}

__global__ __launch_bounds__(256)
void se_scale_elu_kernel(float* __restrict__ x, const float* __restrict__ sc)
{
    const size_t n = (size_t)BB * CC * HWP;
    for (size_t i = (size_t)blockIdx.x * 256 + threadIdx.x; i < n;
         i += (size_t)gridDim.x * 256) {
        int bc = (int)(i / HWP);
        float v = x[i] * sc[bc];
        x[i] = v > 0.f ? v : (__expf(v) - 1.0f);
    }
}

// ---------------------------------------------------------------------------
extern "C" void kernel_launch(void* const* d_in, const int* in_sizes, int n_in,
                              void* d_out, int out_size, void* d_ws, size_t ws_size,
                              hipStream_t stream)
{
    (void)in_sizes; (void)n_in; (void)out_size; (void)ws_size;

    const float* t_feat  = (const float*)d_in[0];
    const float* s_feat  = (const float*)d_in[1];
    const float* directs = (const float*)d_in[2];
    // d_in[3] = image_shape (int, unused)
    const float* disps   = (const float*)d_in[4];
    const float* wq      = (const float*)d_in[5];
    const float* bq      = (const float*)d_in[6];
    const float* wkc     = (const float*)d_in[7];
    const float* bkc     = (const float*)d_in[8];
    const float* metric  = (const float*)d_in[9];
    const float* alpha   = (const float*)d_in[10];
    const float* beta    = (const float*)d_in[11];
    const float* w_cf    = (const float*)d_in[12];
    const float* b_cf    = (const float*)d_in[13];
    const float* w_da    = (const float*)d_in[14];
    const float* b_da    = (const float*)d_in[15];
    const float* w_ex    = (const float*)d_in[16];
    const float* b_ex    = (const float*)d_in[17];
    const float* w_fg    = (const float*)d_in[18];
    const float* b_fg    = (const float*)d_in[19];
    const float* w_rd    = (const float*)d_in[20];
    const float* b_rd    = (const float*)d_in[21];
    const float* w_se1   = (const float*)d_in[22];
    const float* w_se2   = (const float*)d_in[23];

    float* x_out = (float*)d_out;                                  // [B,C,H,W]
    float* P_out = x_out + (size_t)BB * CC * HWP;                  // [B,D,H,W]

    float* ws = (float*)d_ws;
    const size_t SZ_CM = (size_t)BB * CM * HWP;   // 3,932,160
    const size_t SZ_C  = (size_t)BB * CC * HWP;   // 15,728,640
    float* ws_q     = ws;
    float* ws_k     = ws_q  + SZ_CM;
    float* ws_Cv    = ws_k  + SZ_CM;
    float* ws_da    = ws_Cv + SZ_CM;
    float* ws_fused = ws_da + SZ_CM;
    float* ws_y     = ws_fused + SZ_C;
    float* ws_s     = ws_y + BB * CC;
    _Float16* ws_wh = (_Float16*)(ws_s + BB * CC); // 147456 halves, 16B aligned

    // 0: pack conv weights to f16 [O][tap][c]
    wrd_pack_kernel<<<576, 256, 0, stream>>>(w_rd, ws_wh);

    // 1-2: q/k projection + l2norm (WMMA)
    qk_norm_kernel<<<BB * (HWP / 64), 256, 0, stream>>>(t_feat, wq, bq, ws_q);
    qk_norm_kernel<<<BB * (HWP / 64), 256, 0, stream>>>(s_feat, wkc, bkc, ws_k);

    // 3: fused warp + sim/tanh + disparity gate (no warped-volume in memory)
    warp_sim_kernel<<<BB * HH, 320, 0, stream>>>(ws_q, ws_k, metric, disps, directs,
                                                 alpha, beta, w_da, b_da, ws_Cv, ws_da);

    // 4: per-slice 3x3 conv * da, softmax over D -> P (second output)
    cf_softmax_kernel<<<BB * HH, 256, 0, stream>>>(ws_Cv, ws_da, w_cf, b_cf, P_out);

    // 5: expansion GEMM (WMMA) + fusion gate
    cost_fuse_kernel<<<BB * (HWP / 16), 256, 0, stream>>>(P_out, t_feat, w_ex, b_ex,
                                                          w_fg, b_fg, ws_fused);

    // 6: heavy C->C 3x3 conv as implicit GEMM (WMMA)
    conv3x3_wmma_kernel<<<BB * HH * (WW / 32), 256, 0, stream>>>(ws_fused, ws_wh,
                                                                 b_rd, x_out);

    // 7-9: SE layer + ELU, in place on x
    se_reduce_kernel<<<BB * CC, 256, 0, stream>>>(x_out, ws_y);
    se_fc_kernel<<<1, 256, 0, stream>>>(ws_y, w_se1, w_se2, ws_s);
    se_scale_elu_kernel<<<4096, 256, 0, stream>>>(x_out, ws_s);
}
